// Encoder_40956808135002
// MI455X (gfx1250) — compile-verified, matrix operand order checked by reading, not run on hardware
//
#include <hip/hip_runtime.h>
#include <hip/hip_bf16.h>

// ---------------------------------------------------------------------------
// LSTM encoder for MI455X (gfx1250, wave32, WMMA bf16 16x16x32).
// Strategy:
//   1) embed_cast: gather embeddings time-major, cast to bf16.
//   2) cast_weights: W_ih/W_hh -> bf16, bias = b_ih + b_hh.
//   3) gemm_gx: one big WMMA GEMM Gx[32768,4096] = X·W_ihT + bias
//      (all 512 timesteps of the input projection at once),
//      2x4 register tiling per wave: each B-frag feeds 2 WMMAs,
//      each A-frag feeds 4 WMMAs.
//   4) 512x lstm_step: gates = Gx[t] + h·W_hhT via WMMA, fused gate
//      nonlinearities + c/h update. h double-buffered in bf16.
// ---------------------------------------------------------------------------

#define VOCAB 32000
#define EMB   1024
#define HID   1024
#define BATCH 64
#define SEQ   512
#define G4    (4 * HID)     // 4096
#define ROWS  (SEQ * BATCH) // 32768

typedef __attribute__((ext_vector_type(16))) __bf16        v16bf;
typedef __attribute__((ext_vector_type(8)))  float         v8f;
typedef __attribute__((ext_vector_type(4)))  unsigned int  uint32x4;

// --- CDNA5 bf16 A/B fragment loader (row-major source, ld = leading dim) ---
// A-matrix 16x32 per-lane layout (ISA 7.12.2): lane m = lane%16,
// lanes 0-15 hold K = 0..7 and 16..23, lanes 16-31 hold K = 8..15 and 24..31,
// i.e. two contiguous 16-byte chunks per lane.
__device__ __forceinline__ v16bf load_frag_bf16(const __bf16* base, int ld,
                                                int row0, int k0, int lane) {
  const int m   = lane & 15;
  const int klo = (lane < 16) ? 0 : 8;
  const __bf16* p = base + (size_t)(row0 + m) * ld + (k0 + klo);
  union { v16bf v; uint32x4 u[2]; } r;
  r.u[0] = *(const uint32x4*)(p);
  r.u[1] = *(const uint32x4*)(p + 16);
  return r.v;
}

// --- 1) Embedding gather + bf16 cast, time-major rows (t*BATCH + b) --------
__global__ void embed_cast(const int* __restrict__ src,
                           const float* __restrict__ emb,
                           __bf16* __restrict__ Xbf) {
  const int row = blockIdx.x;            // t*BATCH + b
  const int t = row / BATCH, b = row % BATCH;
  const int tok = src[b * SEQ + t];
  const float* e = emb + (size_t)tok * EMB;
  __bf16* o = Xbf + (size_t)row * EMB;
  for (int c = threadIdx.x; c < EMB; c += blockDim.x)
    o[c] = (__bf16)e[c];
}

// --- 2) Weight casts + combined bias ---------------------------------------
__global__ void cast_weights(const float* __restrict__ Wih,
                             const float* __restrict__ Whh,
                             const float* __restrict__ bih,
                             const float* __restrict__ bhh,
                             __bf16* __restrict__ WihB,
                             __bf16* __restrict__ WhhB,
                             float* __restrict__ bias) {
  const size_t i = (size_t)blockIdx.x * blockDim.x + threadIdx.x;
  const size_t n = (size_t)G4 * EMB;
  if (i < n) {
    WihB[i] = (__bf16)Wih[i];
    WhhB[i] = (__bf16)Whh[i];
  }
  if (i < G4) bias[i] = bih[i] + bhh[i];
}

// --- zero h0 (bf16 double buffer) and d_out (h,c fp32) ---------------------
__global__ void init_state(__bf16* __restrict__ hbf, float* __restrict__ out) {
  const int i = blockIdx.x * blockDim.x + threadIdx.x;
  if (i < 2 * BATCH * HID) {
    hbf[i] = (__bf16)0.0f;
    out[i] = 0.0f;
  }
}

// --- 3) Big GEMM: Gx = X·W_ihT + bias --------------------------------------
// Block = 256 threads = 8 waves. Workgroup tile 256(M) x 64(N):
// wave w owns 32 rows (2 M-tiles) at m0 = bx*256 + w*32 and 4 N-tiles.
// Per k-step: 2 A-frag loads + 4 B-frag loads -> 8 WMMAs (B reused 2x, A 4x).
__global__ void gemm_gx(const __bf16* __restrict__ X,
                        const __bf16* __restrict__ W,
                        const float* __restrict__ bias,
                        float* __restrict__ Gx) {
  const int lane = threadIdx.x & 31;
  const int wv   = threadIdx.x >> 5;
  const int m0   = blockIdx.x * 256 + wv * 32;
  const int n0   = blockIdx.y * 64;

  v8f acc[2][4];
#pragma unroll
  for (int i = 0; i < 2; ++i)
#pragma unroll
    for (int j = 0; j < 4; ++j)
#pragma unroll
      for (int r = 0; r < 8; ++r) acc[i][j][r] = 0.0f;

  for (int k0 = 0; k0 < EMB; k0 += 32) {
    const v16bf a0 = load_frag_bf16(X, EMB, m0,      k0, lane);
    const v16bf a1 = load_frag_bf16(X, EMB, m0 + 16, k0, lane);
#pragma unroll
    for (int j = 0; j < 4; ++j) {
      const v16bf b = load_frag_bf16(W, EMB, n0 + 16 * j, k0, lane);
      acc[0][j] = __builtin_amdgcn_wmma_f32_16x16x32_bf16(
          false, a0, false, b, (short)0, acc[0][j], false, false);
      acc[1][j] = __builtin_amdgcn_wmma_f32_16x16x32_bf16(
          false, a1, false, b, (short)0, acc[1][j], false, false);
    }
  }

  // C/D layout: VGPR r holds M = r + 8*(lane/16), N = lane%16.
  const int nl = lane & 15;
  const int mh = (lane >> 4) * 8;
#pragma unroll
  for (int j = 0; j < 4; ++j) {
    const int n = n0 + 16 * j + nl;
    const float bv = bias[n];
#pragma unroll
    for (int i = 0; i < 2; ++i) {
      const int mb = m0 + 16 * i + mh;
#pragma unroll
      for (int r = 0; r < 8; ++r)
        Gx[(size_t)(mb + r) * G4 + n] = acc[i][j][r] + bv;
    }
  }
}

// --- 4) One recurrent step: gates = Gx[t] + h·W_hhT; update c,h ------------
// Grid (64,4): blockIdx.x = hidden col tile (16), blockIdx.y = batch tile (16).
// Block = 128 threads = 4 waves; wave g computes gate g's 16x16 tile.
__global__ void lstm_step(const float* __restrict__ Gxt,   // [64,4096] (+bias)
                          const __bf16* __restrict__ h_in, // [64,1024]
                          const __bf16* __restrict__ Whh,  // [4096,1024]
                          __bf16* __restrict__ h_out_bf,   // [64,1024]
                          float* __restrict__ h_out_f,     // d_out h
                          float* __restrict__ c_state) {   // d_out c
  __shared__ float gsh[4][16][16];

  const int lane = threadIdx.x & 31;
  const int g    = threadIdx.x >> 5;          // gate: i,f,g,o
  const int n0   = blockIdx.x * 16;
  const int m0   = blockIdx.y * 16;
  const int col  = g * HID + n0;              // gate column base in [0,4096)

  const int nl = lane & 15;
  const int mh = (lane >> 4) * 8;

  // Accumulator seeded from precomputed input projection (+both biases).
  v8f acc;
#pragma unroll
  for (int r = 0; r < 8; ++r)
    acc[r] = Gxt[(size_t)(m0 + mh + r) * G4 + col + nl];

  for (int k0 = 0; k0 < HID; k0 += 32) {
    const v16bf a = load_frag_bf16(h_in, HID, m0, k0, lane);
    const v16bf b = load_frag_bf16(Whh, HID, col, k0, lane);
    acc = __builtin_amdgcn_wmma_f32_16x16x32_bf16(
        false, a, false, b, (short)0, acc, false, false);
  }

#pragma unroll
  for (int r = 0; r < 8; ++r)
    gsh[g][mh + r][nl] = acc[r];
  __syncthreads();

  // Elementwise gate math: 128 threads cover 256 (m,n) cells.
  for (int idx = threadIdx.x; idx < 256; idx += 128) {
    const int m = idx >> 4, n = idx & 15;
    float iv = gsh[0][m][n];
    float fv = gsh[1][m][n];
    float gv = gsh[2][m][n];
    float ov = gsh[3][m][n];
    iv = 1.0f / (1.0f + __expf(-iv));
    fv = 1.0f / (1.0f + __expf(-fv));
    gv = tanhf(gv);
    ov = 1.0f / (1.0f + __expf(-ov));
    const size_t off = (size_t)(m0 + m) * HID + (n0 + n);
    const float cv = fv * c_state[off] + iv * gv;
    const float hv = ov * tanhf(cv);
    c_state[off]  = cv;
    h_out_f[off]  = hv;
    h_out_bf[off] = (__bf16)hv;
  }
}

// ---------------------------------------------------------------------------
extern "C" void kernel_launch(void* const* d_in, const int* in_sizes, int n_in,
                              void* d_out, int out_size, void* d_ws, size_t ws_size,
                              hipStream_t stream) {
  const int*   src = (const int*)d_in[0];
  const float* emb = (const float*)d_in[1];
  const float* Wih = (const float*)d_in[2];
  const float* Whh = (const float*)d_in[3];
  const float* bih = (const float*)d_in[4];
  const float* bhh = (const float*)d_in[5];
  float* out = (float*)d_out;          // [2, 64, 1024] -> h then c

  // Workspace layout (~593 MB total).
  char* ws = (char*)d_ws;
  __bf16* Xbf  = (__bf16*)ws; ws += (size_t)ROWS * EMB * 2;      // 64 MB
  __bf16* WihB = (__bf16*)ws; ws += (size_t)G4 * EMB * 2;        //  8 MB
  __bf16* WhhB = (__bf16*)ws; ws += (size_t)G4 * HID * 2;        //  8 MB
  float*  bias = (float*)ws;  ws += (size_t)G4 * 4;              // 16 KB
  __bf16* hbf  = (__bf16*)ws; ws += (size_t)2 * BATCH * HID * 2; // 256 KB
  float*  Gx   = (float*)ws;                                     // 512 MB

  embed_cast<<<ROWS, 256, 0, stream>>>(src, emb, Xbf);
  cast_weights<<<((size_t)G4 * EMB + 255) / 256, 256, 0, stream>>>(
      Wih, Whh, bih, bhh, WihB, WhhB, bias);
  init_state<<<(2 * BATCH * HID + 255) / 256, 256, 0, stream>>>(hbf, out);

  // Gx = X · W_ihT + (b_ih + b_hh): grid 128 x 64, 8 waves/WG, 2x4 tiles/wave.
  gemm_gx<<<dim3(ROWS / 256, G4 / 64), 256, 0, stream>>>(Xbf, WihB, bias, Gx);

  float* h_f = out;                 // final h
  float* c_f = out + BATCH * HID;   // running/final c
  for (int t = 0; t < SEQ; ++t) {
    const float* Gxt  = Gx + (size_t)t * BATCH * G4;
    __bf16*      hin  = hbf + (size_t)(t & 1) * BATCH * HID;
    __bf16*      hout = hbf + (size_t)((t + 1) & 1) * BATCH * HID;
    lstm_step<<<dim3(HID / 16, BATCH / 16), 128, 0, stream>>>(
        Gxt, hin, WhhB, hout, h_f, c_f);
  }
}
